// AttentionEncoder_38826504356695
// MI455X (gfx1250) — matrix operand order repaired and checked by reading, not compiled, not run on hardware
//
#include <hip/hip_runtime.h>
#include <hip/hip_bf16.h>

typedef __bf16 bf16;
typedef __attribute__((ext_vector_type(16))) __bf16 v16bf;
typedef __attribute__((ext_vector_type(8)))  __bf16 bf16x8;
typedef __attribute__((ext_vector_type(8)))  float  v8f;
typedef __attribute__((ext_vector_type(4)))  int    v4i;

#define B_       8
#define N_       961
#define NPAD     976      // 61 * 16
#define NT_      61       // 16-wide tiles covering N
#define PL_      50
#define D_       768
#define H_       12
#define HD_      64
#define MP_      1536
#define RUVC     2304     // 3*D
#define ROWS     (B_*N_)  // 7688
#define ROWS_PAD 7696     // 481 * 16
#define RT_      481
#define CT_      144      // 2304/16

#if __has_builtin(__builtin_amdgcn_global_load_async_to_lds_b128)
#define HAVE_ASYNC_LDS 1
#else
#define HAVE_ASYNC_LDS 0
#endif

// integer -> addrspace pointer casts (low 32 bits of a generic shared address
// are the LDS byte offset; global generic address == AS1 address). The async
// builtin's parameters are v4i* per the clang diagnostic.
#define AS1V4I(p) ((__attribute__((address_space(1))) v4i*)(uintptr_t)(p))
#define AS3V4I(p) ((__attribute__((address_space(3))) v4i*)(uintptr_t)(p))

__device__ __forceinline__ void wait_async0() {
#if __has_builtin(__builtin_amdgcn_s_wait_asynccnt)
  __builtin_amdgcn_s_wait_asynccnt(0);
#else
  asm volatile("s_wait_asynccnt 0x0" ::: "memory");
#endif
}

__device__ __forceinline__ v8f wmma_bf16(v16bf a, v16bf b, v8f c) {
  // D = A(16x32) * B(32x16) + C, f32 accumulate
  return __builtin_amdgcn_wmma_f32_16x16x32_bf16(false, a, false, b, (short)0, c, false, false);
}

// ---------------- weight fusion (one-time, tiny) ----------------

// A0 (51 x 200): rows 0..49 = W1, row 50 = b1  (augmented affine form)
__global__ void k_aug_init(const float* __restrict__ W1, const float* __restrict__ b1,
                           float* __restrict__ A0) {
  int i = blockIdx.x * 256 + threadIdx.x;
  if (i >= 51 * 200) return;
  int r = i / 200, j = i % 200;
  A0[i] = (r < 50) ? W1[r * 200 + j] : b1[j];
}

// out(51 x c) = A(51 x k) @ W(k x c); bias added to augmented row 50
__global__ void k_affine(const float* __restrict__ A, const float* __restrict__ W,
                         const float* __restrict__ bias, float* __restrict__ out,
                         int k, int c) {
  int i = blockIdx.x * 256 + threadIdx.x;
  if (i >= 51 * c) return;
  int r = i / c, j = i % c;
  float acc = (r == 50) ? bias[j] : 0.f;
  const float* arow = A + (size_t)r * k;
  for (int p = 0; p < k; ++p) acc += arow[p] * W[(size_t)p * c + j];
  out[(size_t)r * c + j] = acc;
}

// Wt[j][kk] = bf16(Afinal[kk][j]) for kk<50 else 0   (2304 x 64, K padded)
__global__ void k_wt_cvt(const float* __restrict__ A5, bf16* __restrict__ wt) {
  int i = blockIdx.x * 256 + threadIdx.x;
  if (i >= RUVC * 64) return;
  int j = i / 64, kk = i & 63;
  wt[i] = (kk < PL_) ? (bf16)A5[(size_t)kk * RUVC + j] : (bf16)0.f;
}

// xbf: (7696 x 64) bf16, zero padded in rows and K
__global__ void k_xcvt(const float* __restrict__ x, bf16* __restrict__ xbf) {
  int i = blockIdx.x * 256 + threadIdx.x;
  if (i >= ROWS_PAD * 64) return;
  int row = i / 64, d = i & 63;
  xbf[i] = (row < ROWS && d < PL_) ? (bf16)x[(size_t)row * PL_ + d] : (bf16)0.f;
}

// wp = Wp1 @ Wp2  -> bf16 (NPAD x 64), zero padded rows
__global__ void k_wp(const float* __restrict__ Wp1, const float* __restrict__ Wp2,
                     bf16* __restrict__ wpb) {
  int i = blockIdx.x * 256 + threadIdx.x;
  if (i >= NPAD * 64) return;
  if (i < N_ * 64) {
    const float* row = Wp1 + (size_t)i * MP_;
    float acc = 0.f;
    for (int p = 0; p < MP_; ++p) acc += row[p] * Wp2[p];
    wpb[i] = (bf16)acc;
  } else {
    wpb[i] = (bf16)0.f;
  }
}

__global__ void k_cbias(const float* __restrict__ bp1, const float* __restrict__ Wp2,
                        const float* __restrict__ bp2, float* __restrict__ cb) {
  __shared__ float red[256];
  float acc = 0.f;
  for (int p = threadIdx.x; p < MP_; p += 256) acc += bp1[p] * Wp2[p];
  red[threadIdx.x] = acc;
  __syncthreads();
  for (int s = 128; s > 0; s >>= 1) {
    if (threadIdx.x < (unsigned)s) red[threadIdx.x] += red[threadIdx.x + s];
    __syncthreads();
  }
  if (threadIdx.x == 0) cb[0] = red[0] + bp2[0];
}

__global__ void k_qinit(float* __restrict__ out, const float* __restrict__ cb) {
  int i = blockIdx.x * 128 + threadIdx.x;
  if (i < B_ * H_) out[i] = cb[0];  // q = o.wp + (bp1.Wp2 + bp2)
}

// ---------------- main GEMM: ruv = X(7688x64) @ Wt^T(64x2304) + bias ----------------
__global__ void __launch_bounds__(256)
k_ruv_gemm(const bf16* __restrict__ xbf, const bf16* __restrict__ wt,
           const float* __restrict__ biasrow,
           bf16* __restrict__ rb, bf16* __restrict__ ub, bf16* __restrict__ vb) {
  int wave = threadIdx.x >> 5;
  int lane = threadIdx.x & 31;
  int gtile = blockIdx.x * 8 + wave;
  if (gtile >= RT_ * CT_) return;
  int rt = gtile / CT_, ct = gtile % CT_;
  int hlf = lane >> 4, lr = lane & 15;

  // A fragments (16x32 bf16 each): lane holds row rt*16+lr; K runs split per half
  const bf16x8* xr = (const bf16x8*)(xbf + (size_t)(rt * 16 + lr) * 64);
  bf16x8 c00 = xr[2 * hlf], c01 = xr[2 * hlf + 1], c10 = xr[4 + 2 * hlf], c11 = xr[5 + 2 * hlf];
  v16bf a0, a1;
#pragma unroll
  for (int j = 0; j < 8; ++j) {
    a0[j] = c00[j]; a0[8 + j] = c01[j];
    a1[j] = c10[j]; a1[8 + j] = c11[j];
  }
  // B fragments: lane holds column ct*16+lr; element e -> k = 32w + hlf*16 + e
  int col = ct * 16 + lr;
  const bf16x8* wr = (const bf16x8*)(wt + (size_t)col * 64);
  bf16x8 d00 = wr[2 * hlf], d01 = wr[2 * hlf + 1];
  bf16x8 d10 = wr[4 + 2 * hlf], d11 = wr[5 + 2 * hlf];
  v16bf b0, b1;
#pragma unroll
  for (int j = 0; j < 8; ++j) {
    b0[j] = d00[j]; b0[8 + j] = d01[j];
    b1[j] = d10[j]; b1[8 + j] = d11[j];
  }
  float bias = biasrow[col];
  v8f c;
#pragma unroll
  for (int e = 0; e < 8; ++e) c[e] = bias;
  c = wmma_bf16(a0, b0, c);
  c = wmma_bf16(a1, b1, c);

  // scatter into r/u/v [h][b][n][d] bf16
  int t = col / D_;
  int h = (col % D_) / HD_;
  int d = col & 63;
  bf16* dst = (t == 0) ? rb : (t == 1) ? ub : vb;
#pragma unroll
  for (int e = 0; e < 8; ++e) {
    int mrow = rt * 16 + e + 8 * hlf;
    if (mrow < ROWS) {
      int bb = mrow / N_, n = mrow % N_;
      dst[((size_t)(h * B_ + bb) * NPAD + n) * 64 + d] = (bf16)c[e];
    }
  }
}

// ---------------- fused attention + projection ----------------
// Per wave: 16 query rows. Per m-tile: S = R.U^T and T = Wp.V^T (same fragment
// layout), combined by online softmax: q += sum_n [sum_m e^{S} T] / [sum_m e^{S}]
// u/v tiles double-buffered in LDS; next tile prefetched with async-to-LDS.
__global__ void __launch_bounds__(256)
k_attn(const bf16* __restrict__ rb, const bf16* __restrict__ ub,
       const bf16* __restrict__ vb, const bf16* __restrict__ wpb,
       float* __restrict__ out) {
  __shared__ bf16 uT[2][16 * 64];
  __shared__ bf16 vT[2][16 * 64];
  int hb = blockIdx.x >> 3;        // h*8 + b
  int chunk = blockIdx.x & 7;
  int tid = threadIdx.x;
  int wave = tid >> 5, lane = tid & 31;
  int hlf = lane >> 4, lr = lane & 15;
  int ntile = chunk * 8 + wave;
  bool active = ntile < NT_;
  int nt = active ? ntile : (NT_ - 1);
  int nbase = nt * 16;

  // A fragments for R and Wp (row = nbase+lr)
  const bf16x8* rr = (const bf16x8*)(rb + ((size_t)hb * NPAD + nbase + lr) * 64);
  const bf16x8* wr = (const bf16x8*)(wpb + (size_t)(nbase + lr) * 64);
  v16bf ar0, ar1, aw0, aw1;
  {
    bf16x8 r00 = rr[2 * hlf], r01 = rr[2 * hlf + 1], r10 = rr[4 + 2 * hlf], r11 = rr[5 + 2 * hlf];
    bf16x8 w00 = wr[2 * hlf], w01 = wr[2 * hlf + 1], w10 = wr[4 + 2 * hlf], w11 = wr[5 + 2 * hlf];
#pragma unroll
    for (int j = 0; j < 8; ++j) {
      ar0[j] = r00[j]; ar0[8 + j] = r01[j]; ar1[j] = r10[j]; ar1[8 + j] = r11[j];
      aw0[j] = w00[j]; aw0[8 + j] = w01[j]; aw1[j] = w10[j]; aw1[8 + j] = w11[j];
    }
  }
  float M[8], num[8], den[8];
#pragma unroll
  for (int e = 0; e < 8; ++e) { M[e] = -3.0e38f; num[e] = 0.f; den[e] = 0.f; }

  const uint4* usrc = (const uint4*)(ub + (size_t)hb * NPAD * 64);
  const uint4* vsrc = (const uint4*)(vb + (size_t)hb * NPAD * 64);

  // stage 16x64 bf16 u,v tiles (128 uint4 each): threads 0..127 -> u, 128..255 -> v
  auto issue_tile = [&](int ms, int buf) {
    if (tid < 128) {
#if HAVE_ASYNC_LDS
      __builtin_amdgcn_global_load_async_to_lds_b128(
          AS1V4I(usrc + ms * 128 + tid), AS3V4I(((uint4*)uT[buf]) + tid), 0, 0);
#else
      ((uint4*)uT[buf])[tid] = usrc[ms * 128 + tid];
#endif
    } else {
#if HAVE_ASYNC_LDS
      __builtin_amdgcn_global_load_async_to_lds_b128(
          AS1V4I(vsrc + ms * 128 + (tid - 128)),
          AS3V4I(((uint4*)vT[buf]) + (tid - 128)), 0, 0);
#else
      ((uint4*)vT[buf])[tid - 128] = vsrc[ms * 128 + (tid - 128)];
#endif
    }
  };

  issue_tile(0, 0);

  for (int ms = 0; ms < NT_; ++ms) {
    int buf = ms & 1;
    int mbase = ms * 16;
#if HAVE_ASYNC_LDS
    wait_async0();                 // tile(ms) landed in LDS (per issuing wave)
#endif
    __syncthreads();               // visible to all; prior consumers of buf done
    if (ms + 1 < NT_) issue_tile(ms + 1, buf ^ 1);

    v16bf bu0, bu1, bv0, bv1;
    const bf16x8* ul = (const bf16x8*)uT[buf];
    const bf16x8* vl = (const bf16x8*)vT[buf];
    int bi = lr * 8 + hlf * 2;     // (lr*64 + hlf*16)/8
    bf16x8 u00 = ul[bi], u01 = ul[bi + 1], u10 = ul[bi + 4], u11 = ul[bi + 5];
    bf16x8 v00 = vl[bi], v01 = vl[bi + 1], v10 = vl[bi + 4], v11 = vl[bi + 5];
#pragma unroll
    for (int j = 0; j < 8; ++j) {
      bu0[j] = u00[j]; bu0[8 + j] = u01[j]; bu1[j] = u10[j]; bu1[8 + j] = u11[j];
      bv0[j] = v00[j]; bv0[8 + j] = v01[j]; bv1[j] = v10[j]; bv1[8 + j] = v11[j];
    }
    v8f s, tt;
#pragma unroll
    for (int e = 0; e < 8; ++e) { s[e] = 0.f; tt[e] = 0.f; }
    s  = wmma_bf16(ar0, bu0, s);
    s  = wmma_bf16(ar1, bu1, s);
    tt = wmma_bf16(aw0, bv0, tt);
    tt = wmma_bf16(aw1, bv1, tt);

    // online softmax: only the row max needs a per-tile cross-lane reduction;
    // num/den are kept as per-lane partials (rescale factor is row-uniform).
    bool mvalid = (mbase + lr) < N_;
#pragma unroll
    for (int e = 0; e < 8; ++e) {
      float sv = mvalid ? s[e] : -3.0e38f;
      float mx = sv;
#pragma unroll
      for (int mk = 1; mk < 16; mk <<= 1) mx = fmaxf(mx, __shfl_xor(mx, mk, 32));
      float newM = fmaxf(M[e], mx);
      float sc = __expf(M[e] - newM);
      float p  = mvalid ? __expf(sv - newM) : 0.f;
      den[e] = den[e] * sc + p;
      num[e] = num[e] * sc + p * tt[e];
      M[e] = newM;
    }
  }

  // final cross-lane reduction of per-lane partials (once per wave)
#pragma unroll
  for (int e = 0; e < 8; ++e) {
    float d0 = den[e], n0 = num[e];
#pragma unroll
    for (int mk = 1; mk < 16; mk <<= 1) {
      d0 += __shfl_xor(d0, mk, 32);
      n0 += __shfl_xor(n0, mk, 32);
    }
    den[e] = d0; num[e] = n0;
  }

  if (active && lr == 0) {   // lanes 0 and 16 own rows e+8*hlf
    float acc = 0.f;
#pragma unroll
    for (int e = 0; e < 8; ++e) {
      int n = nbase + e + 8 * hlf;
      if (n < N_) acc += num[e] / den[e];
    }
    int h = hb >> 3, bb = hb & 7;
    atomicAdd(out + bb * H_ + h, acc);
  }
}

// ---------------- launcher ----------------
extern "C" void kernel_launch(void* const* d_in, const int* in_sizes, int n_in,
                              void* d_out, int out_size, void* d_ws, size_t ws_size,
                              hipStream_t stream) {
  const float* x    = (const float*)d_in[0];
  const float* W1   = (const float*)d_in[1];
  const float* b1   = (const float*)d_in[2];
  const float* W2   = (const float*)d_in[3];
  const float* b2   = (const float*)d_in[4];
  const float* W3   = (const float*)d_in[5];
  const float* b3   = (const float*)d_in[6];
  const float* W4   = (const float*)d_in[7];
  const float* b4   = (const float*)d_in[8];
  const float* Wruv = (const float*)d_in[9];
  const float* bruv = (const float*)d_in[10];
  const float* Wp1  = (const float*)d_in[11];
  const float* bp1  = (const float*)d_in[12];
  const float* Wp2  = (const float*)d_in[13];
  const float* bp2  = (const float*)d_in[14];
  float* out = (float*)d_out;

  char* ws = (char*)d_ws;
  size_t off = 0;
  auto alloc = [&](size_t bytes) -> void* {
    void* p = ws + off;
    off = (off + bytes + 255) & ~(size_t)255;
    return p;
  };
  float* A0  = (float*)alloc((size_t)51 * RUVC * 4);
  float* A1  = (float*)alloc((size_t)51 * RUVC * 4);
  bf16*  wt  = (bf16*)alloc((size_t)RUVC * 64 * 2);
  bf16*  xbf = (bf16*)alloc((size_t)ROWS_PAD * 64 * 2);
  bf16*  wpb = (bf16*)alloc((size_t)NPAD * 64 * 2);
  float* cb  = (float*)alloc(256);
  bf16*  rbf = (bf16*)alloc((size_t)H_ * B_ * NPAD * 64 * 2);
  bf16*  ubf = (bf16*)alloc((size_t)H_ * B_ * NPAD * 64 * 2);
  bf16*  vbf = (bf16*)alloc((size_t)H_ * B_ * NPAD * 64 * 2);
  if (ws_size < off) return;  // workspace too small; bail without corruption

  // weight fusion chain (augmented affine): A = [[W],[b]]
  k_aug_init<<<(51 * 200 + 255) / 256, 256, 0, stream>>>(W1, b1, A0);
  k_affine<<<(51 * 200 + 255) / 256, 256, 0, stream>>>(A0, W2, b2, A1, 200, 200);
  k_affine<<<(51 * 200 + 255) / 256, 256, 0, stream>>>(A1, W3, b3, A0, 200, 200);
  k_affine<<<(51 * 768 + 255) / 256, 256, 0, stream>>>(A0, W4, b4, A1, 200, 768);
  k_affine<<<(51 * RUVC + 255) / 256, 256, 0, stream>>>(A1, Wruv, bruv, A0, 768, RUVC);
  k_wt_cvt<<<(RUVC * 64 + 255) / 256, 256, 0, stream>>>(A0, wt);
  k_xcvt<<<(ROWS_PAD * 64 + 255) / 256, 256, 0, stream>>>(x, xbf);
  k_wp<<<(NPAD * 64 + 255) / 256, 256, 0, stream>>>(Wp1, Wp2, wpb);
  k_cbias<<<1, 256, 0, stream>>>(bp1, Wp2, bp2, cb);
  k_qinit<<<1, 128, 0, stream>>>(out, cb);

  // ruv GEMM (WMMA bf16): 481x144 16x16 tiles, 8 waves/block
  k_ruv_gemm<<<(RT_ * CT_) / 8, 256, 0, stream>>>(xbf, wt, A0 + (size_t)50 * RUVC,
                                                  rbf, ubf, vbf);
  // fused attention + projection: 96 (h,b) x 8 row-chunks
  k_attn<<<96 * 8, 256, 0, stream>>>(rbf, ubf, vbf, wpb, out);
}
